// MolecularRetNet_15994458210954
// MI455X (gfx1250) — matrix operand order repaired and checked by reading, not compiled, not run on hardware
//
#include <hip/hip_runtime.h>
#include <hip/hip_bf16.h>

#define BB 32
#define SS 512
#define DD 256
#define HH 8
#define HDIM 32
#define FFND 128
#define LLAY 3
#define BSROWS (BB * SS)

typedef __bf16 bf16_t;
typedef __bf16 v16bf __attribute__((ext_vector_type(16)));
typedef float  v8f   __attribute__((ext_vector_type(8)));
typedef unsigned int v4u __attribute__((ext_vector_type(4)));
typedef unsigned int v8u __attribute__((ext_vector_type(8)));

union BF16x16 { v16bf v; uint4 q[2]; };

__device__ __forceinline__ float gelu_f(float x) {
  return 0.5f * x * (1.0f + erff(x * 0.70710678118654752f));
}

__device__ __forceinline__ void wait_tensorcnt0() {
#if __has_builtin(__builtin_amdgcn_s_wait_tensorcnt)
  __builtin_amdgcn_s_wait_tensorcnt(0);
#else
  asm volatile("s_wait_tensorcnt 0x0" ::: "memory");
#endif
}

// Issue a 2D TDM tile load Global->LDS.
// rows x cols elements (bf16), row stride 'stride' elements in memory,
// LDS rows padded by (padDwMinus1+1) dwords after every 8<<padIntervalLog2 B.
__device__ __forceinline__ void tdm_load_tile_bf16(
    const bf16_t* gptr, unsigned ldsAddr, int cols, int rows,
    int tdim0, int tdim1, int stride, int padIntervalLog2, int padDwMinus1) {
  unsigned long long ga = (unsigned long long)(size_t)gptr;
  v4u g0;
  g0[0] = 1u;                                   // count = 1 valid descriptor
  g0[1] = ldsAddr;                              // LDS byte address
  g0[2] = (unsigned)ga;                         // global addr lo
  g0[3] = (unsigned)(ga >> 32) | (2u << 30);    // global addr hi | type=2
  v8u g1;
  g1[0] = (1u << 16)                            // data_size = 2 bytes
        | (1u << 20)                            // pad_enable
        | ((unsigned)padIntervalLog2 << 22)     // pad_interval
        | ((unsigned)padDwMinus1 << 25);        // pad_amount (DWORDs-1)
  g1[1] = ((unsigned)tdim0 & 0xFFFFu) << 16;                       // tensor_dim0 lo
  g1[2] = ((unsigned)tdim0 >> 16) | (((unsigned)tdim1 & 0xFFFFu) << 16);
  g1[3] = ((unsigned)tdim1 >> 16) | ((unsigned)cols << 16);        // tile_dim0
  g1[4] = (unsigned)rows;                                          // tile_dim1 (tile_dim2=0)
  g1[5] = (unsigned)stride;                                        // tensor_dim0_stride lo
  g1[6] = 0u;
  g1[7] = 0u;
  asm volatile("tensor_load_to_lds %0, %1" :: "s"(g0), "s"(g1) : "memory");
}

// ---------------------------------------------------------------------------
// Embedding + ReLU : X[b,s,:] = relu(emb[v[b,s]]), emb row 0 forced to zero
// ---------------------------------------------------------------------------
__global__ __launch_bounds__(256) void embed_relu_kernel(
    const int* __restrict__ v, const float* __restrict__ emb,
    float* __restrict__ X) {
  int t = blockIdx.x * 256 + threadIdx.x;
  if (t >= BSROWS * DD) return;
  int row = t >> 8;
  int c = t & (DD - 1);
  int idx = v[row];
  float val = (idx == 0) ? 0.0f : fmaxf(emb[(size_t)idx * DD + c], 0.0f);
  X[t] = val;
}

// ---------------------------------------------------------------------------
// f32 -> bf16 weight convert / QKV head-repack ((H,D,HD) -> [D][H*HD])
// ---------------------------------------------------------------------------
__global__ __launch_bounds__(256) void cvt_bf16_kernel(
    const float* __restrict__ in, bf16_t* __restrict__ out, int n) {
  int t = blockIdx.x * 256 + threadIdx.x;
  if (t < n) out[t] = (bf16_t)in[t];
}

__global__ __launch_bounds__(256) void repack_qkv_kernel(
    const float* __restrict__ in, bf16_t* __restrict__ out) {
  int t = blockIdx.x * 256 + threadIdx.x;   // t = d*DD + (h*HD+e)
  if (t >= DD * DD) return;
  int d = t >> 8;
  int c = t & (DD - 1);
  int h = c >> 5;
  int e = c & (HDIM - 1);
  out[t] = (bf16_t)in[((size_t)(h * DD + d)) * HDIM + e];
}

// ---------------------------------------------------------------------------
// LayerNorm over D=256 : one wave per row, bf16 output for WMMA GEMMs
// ---------------------------------------------------------------------------
__global__ __launch_bounds__(256) void layernorm_kernel(
    const float* __restrict__ X, const float* __restrict__ w,
    const float* __restrict__ b, bf16_t* __restrict__ out, int rows) {
  int wave = threadIdx.x >> 5, lane = threadIdx.x & 31;
  int row = blockIdx.x * 8 + wave;
  if (row >= rows) return;
  const float4* xp = (const float4*)(X + (size_t)row * DD) + lane * 2;
  float4 x0 = xp[0], x1 = xp[1];
  float s  = x0.x + x0.y + x0.z + x0.w + x1.x + x1.y + x1.z + x1.w;
  float ss = x0.x*x0.x + x0.y*x0.y + x0.z*x0.z + x0.w*x0.w +
             x1.x*x1.x + x1.y*x1.y + x1.z*x1.z + x1.w*x1.w;
#pragma unroll
  for (int m = 1; m < 32; m <<= 1) {
    s  += __shfl_xor(s,  m, 32);
    ss += __shfl_xor(ss, m, 32);
  }
  float mu = s * (1.0f / DD);
  float var = ss * (1.0f / DD) - mu * mu;
  float rstd = rsqrtf(var + 1e-5f);
  int c = lane * 8;
  float xs[8] = {x0.x, x0.y, x0.z, x0.w, x1.x, x1.y, x1.z, x1.w};
  bf16_t* op = out + (size_t)row * DD + c;
#pragma unroll
  for (int i = 0; i < 8; ++i)
    op[i] = (bf16_t)((xs[i] - mu) * rstd * w[c + i] + b[c + i]);
}

// ---------------------------------------------------------------------------
// xPos rotary-with-scale. transposed=1 writes K^T layout [B,H,HD,S] (bf16)
// ---------------------------------------------------------------------------
__global__ __launch_bounds__(256) void xpos_kernel(
    const float* __restrict__ in, bf16_t* __restrict__ out,
    int downscale, int transposed) {
  int t = blockIdx.x * 256 + threadIdx.x;          // pairs: BS * 128
  if (t >= BSROWS * (DD / 2)) return;
  int row = t >> 7;                                // b*S+s
  int pc = t & 127;
  int h = pc >> 4, p = pc & 15;
  int s = row & (SS - 1);
  int col = h * HDIM + 2 * p;
  float x0 = in[(size_t)row * DD + col];
  float x1 = in[(size_t)row * DD + col + 1];
  float scale = (2.0f * p + 0.4f * HDIM) / (1.4f * HDIM);
  float sm = __powf(scale, (float)s * (1.0f / 512.0f));
  if (downscale) sm = 1.0f / sm;
  float invf = __powf(10000.0f, -(float)p * (1.0f / 16.0f));
  float ang = (float)s * invf;
  float sn = __sinf(ang) * sm, cs = __cosf(ang) * sm;
  float o0 = x0 * cs - x1 * sn;
  float o1 = x1 * cs + x0 * sn;
  if (!transposed) {
    out[(size_t)row * DD + col]     = (bf16_t)o0;
    out[(size_t)row * DD + col + 1] = (bf16_t)o1;
  } else {
    int bb = row >> 9;
    size_t base = ((size_t)(bb * HH + h) * HDIM) * SS;
    out[base + (size_t)(2 * p) * SS + s]     = (bf16_t)o0;
    out[base + (size_t)(2 * p + 1) * SS + s] = (bf16_t)o1;
  }
}

// ---------------------------------------------------------------------------
// Fused causal retention + per-head group norm (flash style, WMMA bf16).
// grid (S/64, B*H), 128 threads (4 waves); each wave owns 16 Q rows.
// Decay gamma^(q-k) applied via multiplicative recurrence (no exp in loop).
// ---------------------------------------------------------------------------
__global__ __launch_bounds__(128) void retention_kernel(
    const bf16_t* __restrict__ Q, const bf16_t* __restrict__ Kt,
    const bf16_t* __restrict__ V, const float* __restrict__ gnw,
    const float* __restrict__ gnb, float* __restrict__ Yn) {
  __shared__ __align__(16) bf16_t sconv[4][16 * 32];
  const int wave = threadIdx.x >> 5;
  const int lane = threadIdx.x & 31;
  const int bh = blockIdx.y;
  const int b = bh >> 3;
  const int h = bh & 7;
  const int qbase = blockIdx.x * 64 + wave * 16;
  const int lm = lane & 15;
  const int kb = (lane < 16) ? 0 : 8;
  const int hi8 = (lane < 16) ? 0 : 8;

  // gamma_h = 1 - exp(linspace(log(1/32), log(1/512), 8))
  float lin = -3.46573590f + (float)h * ((-6.23832463f + 3.46573590f) / 7.0f);
  float lg = __log2f(1.0f - __expf(lin));
  // hoisted decay factors: gamma^(q-k) = rowp * colf, rowp updated per tile
  float cneg0 = exp2f(-lg * (float)lm);
  float cneg1 = cneg0 * exp2f(-16.0f * lg);
  float gstep = exp2f(-32.0f * lg);
  float rowp[8];
#pragma unroll
  for (int j = 0; j < 8; ++j)
    rowp[j] = exp2f(lg * (float)(qbase + j + hi8));
  int qd0 = qbase + hi8;   // (qrow - t0) for j = 0

  BF16x16 qa;
  {
    const bf16_t* qp = Q + ((size_t)(b * SS + qbase + lm)) * DD + h * HDIM;
    qa.q[0] = *(const uint4*)(qp + kb);
    qa.q[1] = *(const uint4*)(qp + 16 + kb);
  }

  v8f y0 = {0.f,0.f,0.f,0.f,0.f,0.f,0.f,0.f};
  v8f y1 = y0;
  bf16_t* sw = sconv[wave];
  const int qhi = qbase + 15;

  for (int t0 = 0; t0 <= qhi; t0 += 32) {
    BF16x16 k0f, k1f;
    const bf16_t* kp = Kt + ((size_t)bh * HDIM) * SS + t0 + lm;
#pragma unroll
    for (int s = 0; s < 16; ++s) {
      int kk = (s < 8) ? (kb + s) : (16 + kb + s - 8);
      k0f.v[s] = kp[(size_t)kk * SS];
      k1f.v[s] = kp[(size_t)kk * SS + 16];
    }

    v8f zc = {0.f,0.f,0.f,0.f,0.f,0.f,0.f,0.f};
    v8f s0 = __builtin_amdgcn_wmma_f32_16x16x32_bf16(false, qa.v, false, k0f.v,
                                                     (short)0, zc, false, false);
    v8f s1 = __builtin_amdgcn_wmma_f32_16x16x32_bf16(false, qa.v, false, k1f.v,
                                                     (short)0, zc, false, false);
#pragma unroll
    for (int j = 0; j < 8; ++j) {
      float d = rowp[j];
      s0[j] *= (qd0 + j >= lm)      ? d * cneg0 : 0.0f;
      s1[j] *= (qd0 + j - 16 >= lm) ? d * cneg1 : 0.0f;
      rowp[j] = d * gstep;
    }
    // D-layout -> A-layout repack of the 16x32 score tile via per-wave LDS
#pragma unroll
    for (int j = 0; j < 8; ++j) {
      int m = j + hi8;
      sw[m * 32 + lm]      = (bf16_t)s0[j];
      sw[m * 32 + 16 + lm] = (bf16_t)s1[j];
    }
    __builtin_amdgcn_wave_barrier();
    asm volatile("s_wait_dscnt 0" ::: "memory");
    BF16x16 af;
    af.q[0] = *(const uint4*)(sw + lm * 32 + kb);
    af.q[1] = *(const uint4*)(sw + lm * 32 + 16 + kb);
    __builtin_amdgcn_wave_barrier();

    const bf16_t* vp = V + ((size_t)(b * SS + t0)) * DD + h * HDIM + lm;
    BF16x16 v0f, v1f;
#pragma unroll
    for (int s = 0; s < 16; ++s) {
      int kk = (s < 8) ? (kb + s) : (16 + kb + s - 8);
      v0f.v[s] = vp[(size_t)kk * DD];
      v1f.v[s] = vp[(size_t)kk * DD + 16];
    }

    y0 = __builtin_amdgcn_wmma_f32_16x16x32_bf16(false, af.v, false, v0f.v,
                                                 (short)0, y0, false, false);
    y1 = __builtin_amdgcn_wmma_f32_16x16x32_bf16(false, af.v, false, v1f.v,
                                                 (short)0, y1, false, false);
    qd0 -= 32;
  }

  // fused per-head (HD=32) group norm + store Yn f32 [B,S,D]
  float gw0 = gnw[h * HDIM + lm],      gb0 = gnb[h * HDIM + lm];
  float gw1 = gnw[h * HDIM + 16 + lm], gb1 = gnb[h * HDIM + 16 + lm];
#pragma unroll
  for (int j = 0; j < 8; ++j) {
    float s  = y0[j] + y1[j];
    float ss = y0[j] * y0[j] + y1[j] * y1[j];
#pragma unroll
    for (int m = 1; m < 16; m <<= 1) {
      s  += __shfl_xor(s,  m, 32);
      ss += __shfl_xor(ss, m, 32);
    }
    float mu = s * (1.0f / HDIM);
    float var = ss * (1.0f / HDIM) - mu * mu;
    float rstd = rsqrtf(var + 1e-5f);
    int qrow = qbase + j + hi8;
    float* op = Yn + ((size_t)(b * SS + qrow)) * DD + h * HDIM;
    op[lm]      = (y0[j] - mu) * rstd * gw0 + gb0;
    op[16 + lm] = (y1[j] - mu) * rstd * gw1 + gb1;
  }
}

// ---------------------------------------------------------------------------
// Gate: out = silu(G) * Yn  (bf16 output feeds WO GEMM)
// ---------------------------------------------------------------------------
__global__ __launch_bounds__(256) void gate_kernel(
    const float* __restrict__ G, const float* __restrict__ Yn,
    bf16_t* __restrict__ out) {
  int t = blockIdx.x * 256 + threadIdx.x;
  if (t >= BSROWS * DD) return;
  float g = G[t];
  out[t] = (bf16_t)(g * (1.0f / (1.0f + __expf(-g))) * Yn[t]);
}

// ---------------------------------------------------------------------------
// Tiled bf16 WMMA GEMM: C = act(A[MxK] * W[KxN] + bias) + resid
// block = 64x64 (128 threads, 4 waves of 32x32), K step 32.
// Tiles are staged into padded LDS by the Tensor Data Mover (TDM):
//   A tile 64x32 -> As[64][40]  (row 64B + 16B pad : interval=3 amount=3)
//   B tile 32x64 -> Bs[32][72]  (row 128B + 16B pad: interval=4 amount=3)
// ---------------------------------------------------------------------------
template <int ACT>  // 0 = none, 1 = exact gelu
__global__ __launch_bounds__(128) void gemm_bf16_kernel(
    const bf16_t* __restrict__ A, const bf16_t* __restrict__ Bw,
    const float* __restrict__ bias, const float* __restrict__ resid,
    float* __restrict__ outF, bf16_t* __restrict__ outB,
    int M, int N, int K) {
  __shared__ __align__(16) bf16_t As[64][40];
  __shared__ __align__(16) bf16_t Bs[32][72];
  const int tid = threadIdx.x;
  const int lane = tid & 31;
  const int wave = tid >> 5;
  const int wm = wave >> 1, wn = wave & 1;
  const int m0 = blockIdx.y * 64;
  const int n0 = blockIdx.x * 64;
  const int lm = lane & 15;
  const int kb = (lane < 16) ? 0 : 8;
  const int hi8 = (lane < 16) ? 0 : 8;

  v8f z = {0.f,0.f,0.f,0.f,0.f,0.f,0.f,0.f};
  v8f acc00 = z, acc01 = z, acc10 = z, acc11 = z;

  const unsigned ldsA = (unsigned)(size_t)&As[0][0];
  const unsigned ldsB = (unsigned)(size_t)&Bs[0][0];

  for (int k0 = 0; k0 < K; k0 += 32) {
    __syncthreads();
    if (wave == 0) {
      tdm_load_tile_bf16(A + (size_t)m0 * K + k0, ldsA,
                         /*cols*/32, /*rows*/64, /*tdim0*/K, /*tdim1*/M,
                         /*stride*/K, /*padIntLog2*/3, /*padDw-1*/3);
      tdm_load_tile_bf16(Bw + (size_t)k0 * N + n0, ldsB,
                         /*cols*/64, /*rows*/32, /*tdim0*/N, /*tdim1*/K,
                         /*stride*/N, /*padIntLog2*/4, /*padDw-1*/3);
      wait_tensorcnt0();
    } else if (wave == 1 && k0 + 32 < K) {
      __builtin_prefetch(A + (size_t)(m0 + lane) * K + k0 + 32, 0, 2);
      __builtin_prefetch(Bw + (size_t)(k0 + 32 + lane) * N + n0, 0, 2);
    }
    __syncthreads();

    BF16x16 a0, a1, b0, b1;
    int ma = wm * 32 + lm;
    a0.q[0] = *(const uint4*)&As[ma][kb];
    a0.q[1] = *(const uint4*)&As[ma][16 + kb];
    a1.q[0] = *(const uint4*)&As[ma + 16][kb];
    a1.q[1] = *(const uint4*)&As[ma + 16][16 + kb];
    int nbase = wn * 32 + lm;
#pragma unroll
    for (int s = 0; s < 16; ++s) {
      int kk = (s < 8) ? (kb + s) : (16 + kb + s - 8);
      b0.v[s] = Bs[kk][nbase];
      b1.v[s] = Bs[kk][nbase + 16];
    }

    acc00 = __builtin_amdgcn_wmma_f32_16x16x32_bf16(false, a0.v, false, b0.v,
                                                    (short)0, acc00, false, false);
    acc01 = __builtin_amdgcn_wmma_f32_16x16x32_bf16(false, a0.v, false, b1.v,
                                                    (short)0, acc01, false, false);
    acc10 = __builtin_amdgcn_wmma_f32_16x16x32_bf16(false, a1.v, false, b0.v,
                                                    (short)0, acc10, false, false);
    acc11 = __builtin_amdgcn_wmma_f32_16x16x32_bf16(false, a1.v, false, b1.v,
                                                    (short)0, acc11, false, false);
  }

  v8f accs[2][2] = {{acc00, acc01}, {acc10, acc11}};
#pragma unroll
  for (int mi = 0; mi < 2; ++mi) {
#pragma unroll
    for (int ni = 0; ni < 2; ++ni) {
      int ng = n0 + wn * 32 + ni * 16 + lm;
      float bv = bias ? bias[ng] : 0.0f;
#pragma unroll
      for (int j = 0; j < 8; ++j) {
        int mg = m0 + wm * 32 + mi * 16 + j + hi8;
        float val = accs[mi][ni][j] + bv;
        if (ACT == 1) val = gelu_f(val);
        if (resid) val += resid[(size_t)mg * N + ng];
        if (outF) outF[(size_t)mg * N + ng] = val;
        if (outB) outB[(size_t)mg * N + ng] = (bf16_t)val;
      }
    }
  }
}

// ---------------------------------------------------------------------------
// Host orchestration
// ---------------------------------------------------------------------------
extern "C" void kernel_launch(void* const* d_in, const int* in_sizes, int n_in,
                              void* d_out, int out_size, void* d_ws,
                              size_t ws_size, hipStream_t stream) {
  const int*   v    = (const int*)d_in[0];
  const float* emb  = (const float*)d_in[1];
  const float* WQ   = (const float*)d_in[2];
  const float* WK   = (const float*)d_in[3];
  const float* WV   = (const float*)d_in[4];
  const float* WG   = (const float*)d_in[5];
  const float* WO   = (const float*)d_in[6];
  const float* gnw  = (const float*)d_in[7];
  const float* gnb  = (const float*)d_in[8];
  const float* ln1w = (const float*)d_in[9];
  const float* ln1b = (const float*)d_in[10];
  const float* ln2w = (const float*)d_in[11];
  const float* ln2b = (const float*)d_in[12];
  const float* w1   = (const float*)d_in[13];
  const float* b1   = (const float*)d_in[14];
  const float* w2   = (const float*)d_in[15];
  const float* b2   = (const float*)d_in[16];

  char* ws = (char*)d_ws;
  size_t off = 0;
  auto alloc = [&](size_t bytes) -> void* {
    void* p = ws + off;
    off += (bytes + 255) & ~(size_t)255;
    return p;
  };
  const size_t nf = (size_t)BSROWS * DD;
  float*  F0  = (float*)alloc(nf * 4);      // X
  float*  F1  = (float*)alloc(nf * 4);      // Y (residual)
  float*  F2  = (float*)alloc(nf * 4);      // Qf32 / Yn
  float*  F3  = (float*)alloc(nf * 4);      // Kf32 / G
  bf16_t* BF0 = (bf16_t*)alloc(nf * 2);     // layernorm out
  bf16_t* BF1 = (bf16_t*)alloc(nf * 2);     // Qbf / gated / FFN hidden
  bf16_t* BF2 = (bf16_t*)alloc(nf * 2);     // Kt bf
  bf16_t* BF3 = (bf16_t*)alloc(nf * 2);     // V bf
  bf16_t* Wq = (bf16_t*)alloc((size_t)DD * DD * 2);
  bf16_t* Wk = (bf16_t*)alloc((size_t)DD * DD * 2);
  bf16_t* Wv = (bf16_t*)alloc((size_t)DD * DD * 2);
  bf16_t* Wg = (bf16_t*)alloc((size_t)DD * DD * 2);
  bf16_t* Wo = (bf16_t*)alloc((size_t)DD * DD * 2);
  bf16_t* W1 = (bf16_t*)alloc((size_t)DD * FFND * 2);
  bf16_t* W2 = (bf16_t*)alloc((size_t)FFND * DD * 2);

  const int thr = 256;
  embed_relu_kernel<<<(BSROWS * DD) / thr, thr, 0, stream>>>(v, emb, F0);

  for (int i = 0; i < LLAY; ++i) {
    const size_t oQ = (size_t)i * HH * DD * HDIM;
    const size_t oD2 = (size_t)i * DD * DD;
    const size_t oFF = (size_t)i * DD * FFND;
    repack_qkv_kernel<<<(DD * DD) / thr, thr, 0, stream>>>(WQ + oQ, Wq);
    repack_qkv_kernel<<<(DD * DD) / thr, thr, 0, stream>>>(WK + oQ, Wk);
    repack_qkv_kernel<<<(DD * DD) / thr, thr, 0, stream>>>(WV + oQ, Wv);
    cvt_bf16_kernel<<<(DD * DD) / thr, thr, 0, stream>>>(WG + oD2, Wg, DD * DD);
    cvt_bf16_kernel<<<(DD * DD) / thr, thr, 0, stream>>>(WO + oD2, Wo, DD * DD);
    cvt_bf16_kernel<<<(DD * FFND) / thr, thr, 0, stream>>>(w1 + oFF, W1, DD * FFND);
    cvt_bf16_kernel<<<(FFND * DD) / thr, thr, 0, stream>>>(w2 + oFF, W2, FFND * DD);

    layernorm_kernel<<<BSROWS / 8, thr, 0, stream>>>(F0, ln1w + i * DD,
                                                     ln1b + i * DD, BF0, BSROWS);
    dim3 gD(DD / 64, BSROWS / 64);
    gemm_bf16_kernel<0><<<gD, 128, 0, stream>>>(BF0, Wq, nullptr, nullptr,
                                                F2, nullptr, BSROWS, DD, DD);
    gemm_bf16_kernel<0><<<gD, 128, 0, stream>>>(BF0, Wk, nullptr, nullptr,
                                                F3, nullptr, BSROWS, DD, DD);
    gemm_bf16_kernel<0><<<gD, 128, 0, stream>>>(BF0, Wv, nullptr, nullptr,
                                                nullptr, BF3, BSROWS, DD, DD);
    xpos_kernel<<<(BSROWS * (DD / 2)) / thr, thr, 0, stream>>>(F2, BF1, 0, 0);
    xpos_kernel<<<(BSROWS * (DD / 2)) / thr, thr, 0, stream>>>(F3, BF2, 1, 1);
    retention_kernel<<<dim3(SS / 64, BB * HH), 128, 0, stream>>>(
        BF1, BF2, BF3, gnw + i * DD, gnb + i * DD, F2);
    gemm_bf16_kernel<0><<<gD, 128, 0, stream>>>(BF0, Wg, nullptr, nullptr,
                                                F3, nullptr, BSROWS, DD, DD);
    gate_kernel<<<(BSROWS * DD) / thr, thr, 0, stream>>>(F3, F2, BF1);
    gemm_bf16_kernel<0><<<gD, 128, 0, stream>>>(BF1, Wo, nullptr, F0,
                                                F1, nullptr, BSROWS, DD, DD);
    layernorm_kernel<<<BSROWS / 8, thr, 0, stream>>>(F1, ln2w + i * DD,
                                                     ln2b + i * DD, BF0, BSROWS);
    dim3 gF1(FFND / 64, BSROWS / 64);
    gemm_bf16_kernel<1><<<gF1, 128, 0, stream>>>(BF0, W1, b1 + i * FFND, nullptr,
                                                 nullptr, BF1, BSROWS, FFND, DD);
    float* outX = (i == LLAY - 1) ? (float*)d_out : F0;
    gemm_bf16_kernel<0><<<gD, 128, 0, stream>>>(BF1, W2, b2 + i * DD, F1,
                                                outX, nullptr, BSROWS, DD, FFND);
  }
}
